// GCN_90838558310850
// MI455X (gfx1250) — matrix operand order repaired and checked by reading, not compiled, not run on hardware
//
#include <hip/hip_runtime.h>

// Problem constants (from reference setup_inputs)
#define BB 8
#define NN 10000
#define EE 160000
#define FF 128
#define HH 256
constexpr int BN = BB * NN;          // 80000 rows; 80000 = 625 * 128 (M-blocks of 8x16)

typedef __attribute__((ext_vector_type(2)))  float  v2f;
typedef __attribute__((ext_vector_type(8)))  float  v8f;
typedef __attribute__((ext_vector_type(16))) __bf16 v16bf;

// ---------------- bf16 helpers (round-to-nearest-even) ----------------
__device__ inline unsigned short f32_to_bf16(float x) {
  unsigned int u = __float_as_uint(x);
  u += 0x7FFFu + ((u >> 16) & 1u);
  return (unsigned short)(u >> 16);
}
__device__ inline float bf16_to_f32(unsigned short h) {
  return __uint_as_float(((unsigned int)h) << 16);
}

__device__ inline v8f wmma_bf16(v16bf a, v16bf b, v8f c) {
  return __builtin_amdgcn_wmma_f32_16x16x32_bf16(false, a, false, b,
                                                 (short)0, c, false, false);
}

// A-matrix 16x32 bf16 fragment (ISA 7.12.2). Per lane the fragment is two
// contiguous 16-byte runs: k = k0+8*half+[0..7] and k0+16+8*half+[0..7].
__device__ inline v16bf load_a_bf16(const unsigned short* __restrict__ A,
                                    int lda, int row0, int k0, int lane) {
  const int m = lane & 15, half = lane >> 4;
  const unsigned short* base = A + (size_t)(row0 + m) * lda + k0 + 8 * half;
  union { v16bf v; uint4 q[2]; } f;
  f.q[0] = *(const uint4*)(base);        // global_load_b128
  f.q[1] = *(const uint4*)(base + 16);   // global_load_b128
  return f.v;
}

// B fragment from LDS (packed layout: frag*256 dwords, lane*8 dwords each)
__device__ inline v16bf lds_b_frag(const unsigned int* p, int frag, int lane) {
  union { v16bf v; uint4 q[2]; } f;
  const uint4* s = (const uint4*)(p + frag * 256 + lane * 8);
  f.q[0] = s[0];                         // ds_load_b128
  f.q[1] = s[1];
  return f.v;
}

// ---------------- weight pre-pack kernels (run once per call, tiny) ----------------
// f32 B fragments for 16x16x4: frag = kt*(Nc/16)+nt, 64 floats per frag,
// lane L stores {W[kt*4+2*half, col], W[kt*4+2*half+1, col]} at pk[frag*64 + L*2].
__global__ __launch_bounds__(256)
void k_pack_b_f32(const float* __restrict__ W, float* __restrict__ pk,
                  int K, int Nc) {
  int i = blockIdx.x * 256 + threadIdx.x;
  int total = (K >> 2) * (Nc >> 4) * 64;
  if (i >= total) return;
  int d = i & 1, lane = (i >> 1) & 31, frag = i >> 6;
  int ntiles = Nc >> 4;
  int kt = frag / ntiles, nt = frag - kt * ntiles;
  int n = lane & 15, half = lane >> 4;
  pk[i] = W[(size_t)(kt * 4 + 2 * half + d) * Nc + nt * 16 + n];
}

// bf16 split+pack for 16x16x32: frag = kt*(Nc/16)+nt, 256 dwords per frag,
// dword v of lane L packs rows k=kt*32+8*half+2*(v&3)+16*(v>>2), +1 at column nt*16+n.
__global__ __launch_bounds__(256)
void k_pack_w_bf16(const float* __restrict__ W, unsigned int* __restrict__ pkHi,
                   unsigned int* __restrict__ pkLo, int K, int Nc) {
  int i = blockIdx.x * 256 + threadIdx.x;
  int total = (K >> 5) * (Nc >> 4) * 256;
  if (i >= total) return;
  int v = i & 7, lane = (i >> 3) & 31, frag = i >> 8;
  int ntiles = Nc >> 4;
  int kt = frag / ntiles, nt = frag - kt * ntiles;
  int n = lane & 15, half = lane >> 4;
  int k = kt * 32 + 8 * half + 2 * (v & 3) + 16 * (v >> 2);
  int col = nt * 16 + n;
  float w0 = W[(size_t)k * Nc + col];
  float w1 = W[(size_t)(k + 1) * Nc + col];
  unsigned short h0 = f32_to_bf16(w0);
  unsigned short l0 = f32_to_bf16(w0 - bf16_to_f32(h0));
  unsigned short h1 = f32_to_bf16(w1);
  unsigned short l1 = f32_to_bf16(w1 - bf16_to_f32(h1));
  pkHi[i] = (unsigned int)h0 | ((unsigned int)h1 << 16);
  pkLo[i] = (unsigned int)l0 | ((unsigned int)l1 << 16);
}

// ---------------- graph setup kernels ----------------
__global__ __launch_bounds__(256)
void k_degree(const long long* __restrict__ ei, unsigned int* __restrict__ deg) {
  int i = blockIdx.x * 256 + threadIdx.x;
  if (i >= BB * EE) return;
  int b = i / EE, e = i - b * EE;
  int dst = (int)ei[(size_t)b * 2 * EE + EE + e];
  atomicAdd(&deg[b * NN + dst], 1u);
}

__global__ __launch_bounds__(256)
void k_dinv(const unsigned int* __restrict__ deg, float* __restrict__ dinv) {
  int i = blockIdx.x * 256 + threadIdx.x;
  if (i >= BN) return;
  dinv[i] = rsqrtf((float)deg[i] + 1.0f);  // +1 = self loop
}

// ---------------- GEMM1: exact fp32 WMMA (16x16x4), B staged in LDS ----------------
// Block = 8 waves = 8 consecutive 16-row tiles, all on one 64-col group.
__global__ __launch_bounds__(256)
void k_gemm_f32wmma(const float* __restrict__ A, const float* __restrict__ Bpk,
                    float* __restrict__ out, int M, int Nc) {
  constexpr int K = 128, KT = K / 4;                 // 32 k-steps
  __shared__ float ldsB[KT * 4 * 64];                // 32 KB
  const int ngrp = Nc >> 6;
  const int mblk = blockIdx.x / ngrp;
  const int ng   = blockIdx.x - mblk * ngrp;
  const int lane = threadIdx.x & 31;
  const int w    = threadIdx.x >> 5;
  const int ntilesTot = Nc >> 4;

  // cooperative copy of this block's packed B slice (frag (kt, ng*4+t))
  for (int q = threadIdx.x; q < KT * 4 * 16; q += 256) {
    int fr = q >> 4, qi = q & 15;
    int kt = fr >> 2, t = fr & 3;
    const float4* src =
        (const float4*)(Bpk + (size_t)(kt * ntilesTot + ng * 4 + t) * 64) + qi;
    ((float4*)ldsB)[q] = *src;
  }
  __syncthreads();

  const int row0 = (mblk * 8 + w) * 16;
  if (row0 >= M) return;
  const int col0 = ng * 64;
  const int m = lane & 15, half = lane >> 4, n = lane & 15;

  const v8f z = {0.f, 0.f, 0.f, 0.f, 0.f, 0.f, 0.f, 0.f};
  v8f acc[4] = {z, z, z, z};

  const float* arow = A + (size_t)(row0 + m) * K + 2 * half;
  for (int kt = 0; kt < KT; ++kt) {
    v2f a = *(const v2f*)(arow + kt * 4);            // global_load_b64
#pragma unroll
    for (int t = 0; t < 4; ++t) {
      v2f b = *(const v2f*)(ldsB + (kt * 4 + t) * 64 + lane * 2);  // ds_load_b64
      acc[t] = __builtin_amdgcn_wmma_f32_16x16x4_f32(false, a, false, b,
                                                     (short)0, acc[t],
                                                     false, false);
    }
  }
#pragma unroll
  for (int t = 0; t < 4; ++t)
#pragma unroll
    for (int v = 0; v < 8; ++v)
      out[(size_t)(row0 + v + 8 * half) * Nc + col0 + 16 * t + n] = acc[t][v];
}

// ---------------- edge aggregation: one wave per edge ----------------
__global__ __launch_bounds__(256)
void k_aggregate(const long long* __restrict__ ei, const float* __restrict__ xw,
                 const float* __restrict__ dinv, float* __restrict__ agg) {
  const int lane = threadIdx.x & 31;
  long long edge = (long long)blockIdx.x * 8 + (threadIdx.x >> 5);
  if (edge >= (long long)BB * EE) return;
  int b = (int)(edge / EE), e = (int)(edge - (long long)b * EE);
  const long long* eb = ei + (size_t)b * 2 * EE;
  __builtin_prefetch(eb + e + 1024, 0, 0);  // global_prefetch_b8 on edge stream
  int src = (int)eb[e];
  int dst = (int)eb[EE + e];
  float nrm = dinv[b * NN + src] * dinv[b * NN + dst];
  const float4 xv = *(const float4*)(xw + (size_t)(b * NN + src) * FF + lane * 4);
  float* dp = agg + (size_t)(b * NN + dst) * FF + lane * 4;
  unsafeAtomicAdd(dp + 0, nrm * xv.x);
  unsafeAtomicAdd(dp + 1, nrm * xv.y);
  unsafeAtomicAdd(dp + 2, nrm * xv.z);
  unsafeAtomicAdd(dp + 3, nrm * xv.w);
}

// ---------------- h = relu(agg + dii*xw + bc) + x0, split to bf16 ----------------
__global__ __launch_bounds__(256)
void k_finalize(const float* __restrict__ agg, const float* __restrict__ xw,
                const float* __restrict__ dinv, const float* __restrict__ bc,
                const float* __restrict__ x0, unsigned short* __restrict__ hhi,
                unsigned short* __restrict__ hlo) {
  size_t i = (size_t)blockIdx.x * 256 + threadIdx.x;
  if (i >= (size_t)BN * FF) return;
  int row = (int)(i >> 7);
  int f = (int)(i & 127);
  float di = dinv[row];
  float conv = agg[i] + di * di * xw[i] + bc[f];
  float r = conv > 0.f ? conv : 0.f;
  float h = r + x0[i];
  unsigned short hh = f32_to_bf16(h);
  hhi[i] = hh;
  hlo[i] = f32_to_bf16(h - bf16_to_f32(hh));
}

// ---------------- bf16x3 split-precision GEMM (16x16x32), B staged in LDS ----------
// EPI==1: out = leaky_relu(A@W + bias) as fp32
// EPI==2: out = leaky_relu(A@W + bias) stored as bf16 hi/lo split
// KT = K/32. Block = 8 waves = 8 consecutive 16-row tiles on one 64-col group.
template <int EPI, int KT>
__global__ __launch_bounds__(256)
void k_gemm_bf16x3(const unsigned short* __restrict__ Ahi,
                   const unsigned short* __restrict__ Alo,
                   const unsigned int* __restrict__ BhiPk,
                   const unsigned int* __restrict__ BloPk,
                   const float* __restrict__ bias,
                   float* __restrict__ outF,
                   unsigned short* __restrict__ outHi,
                   unsigned short* __restrict__ outLo,
                   int M, int Nc) {
  constexpr int K = KT * 32;
  __shared__ unsigned int ldsB[2 * KT * 4 * 256];    // 32 KB (KT=4) / 64 KB (KT=8)
  const int ngrp = Nc >> 6;
  const int mblk = blockIdx.x / ngrp;
  const int ng   = blockIdx.x - mblk * ngrp;
  const int lane = threadIdx.x & 31;
  const int w    = threadIdx.x >> 5;
  const int ntilesTot = Nc >> 4;
  unsigned int* ldsHi = ldsB;
  unsigned int* ldsLo = ldsB + KT * 4 * 256;

  for (int q = threadIdx.x; q < KT * 4 * 64; q += 256) {
    int fr = q >> 6, qi = q & 63;
    int kt = fr >> 2, t = fr & 3;
    size_t g = (size_t)(kt * ntilesTot + ng * 4 + t) * 256;
    ((uint4*)ldsHi)[q] = ((const uint4*)(BhiPk + g))[qi];
    ((uint4*)ldsLo)[q] = ((const uint4*)(BloPk + g))[qi];
  }
  __syncthreads();

  const int row0 = (mblk * 8 + w) * 16;
  if (row0 >= M) return;
  const int col0 = ng * 64;
  const int half = lane >> 4, n = lane & 15;

  const v8f z = {0.f, 0.f, 0.f, 0.f, 0.f, 0.f, 0.f, 0.f};
  v8f acc[4] = {z, z, z, z};

#pragma unroll
  for (int kt = 0; kt < KT; ++kt) {
    v16bf ahi = load_a_bf16(Ahi, K, row0, kt * 32, lane);
    v16bf alo = load_a_bf16(Alo, K, row0, kt * 32, lane);
#pragma unroll
    for (int t = 0; t < 4; ++t) {
      v16bf bhi = lds_b_frag(ldsHi, kt * 4 + t, lane);
      v16bf blo = lds_b_frag(ldsLo, kt * 4 + t, lane);
      acc[t] = wmma_bf16(ahi, bhi, acc[t]);   // hi*hi
      acc[t] = wmma_bf16(ahi, blo, acc[t]);   // hi*lo
      acc[t] = wmma_bf16(alo, bhi, acc[t]);   // lo*hi
    }
  }

#pragma unroll
  for (int t = 0; t < 4; ++t) {
    int col = col0 + 16 * t + n;
    float bv = bias[col];
#pragma unroll
    for (int v = 0; v < 8; ++v) {
      size_t idx = (size_t)(row0 + v + 8 * half) * Nc + col;
      float y = acc[t][v] + bv;
      y = y > 0.f ? y : 0.01f * y;  // leaky_relu
      if (EPI == 1) {
        outF[idx] = y;
      } else {
        unsigned short h = f32_to_bf16(y);
        outHi[idx] = h;
        outLo[idx] = f32_to_bf16(y - bf16_to_f32(h));
      }
    }
  }
}

// ---------------- launcher ----------------
extern "C" void kernel_launch(void* const* d_in, const int* in_sizes, int n_in,
                              void* d_out, int out_size, void* d_ws, size_t ws_size,
                              hipStream_t stream) {
  (void)in_sizes; (void)n_in; (void)out_size; (void)ws_size;
  const float*     x  = (const float*)d_in[0];        // [B,N,F]
  const long long* ei = (const long long*)d_in[1];    // [B,2,E] int64
  const float*     Wc = (const float*)d_in[2];        // [F,F]
  const float*     bc = (const float*)d_in[3];        // [F]
  const float*     W1 = (const float*)d_in[4];        // [F,H]
  const float*     b1 = (const float*)d_in[5];        // [H]
  const float*     W2 = (const float*)d_in[6];        // [H,H]
  const float*     b2 = (const float*)d_in[7];        // [H]
  float* out = (float*)d_out;                         // [B,N,H]

  // workspace layout (256B aligned)
  char* ws = (char*)d_ws;
  size_t off = 0;
  auto take = [&](size_t bytes) -> char* {
    char* p = ws + off;
    off += (bytes + 255) & ~(size_t)255;
    return p;
  };
  unsigned int*   deg   = (unsigned int*)  take((size_t)BN * 4);
  float*          dinv  = (float*)         take((size_t)BN * 4);
  float*          xw    = (float*)         take((size_t)BN * FF * 4);   // 40.96 MB
  float*          agg   = (float*)         take((size_t)BN * FF * 4);   // 40.96 MB
  float*          WcPk  = (float*)         take((size_t)(FF/4)*(FF/16)*64 * 4);
  unsigned int*   W1pkH = (unsigned int*)  take((size_t)(FF/32)*(HH/16)*256 * 4);
  unsigned int*   W1pkL = (unsigned int*)  take((size_t)(FF/32)*(HH/16)*256 * 4);
  unsigned int*   W2pkH = (unsigned int*)  take((size_t)(HH/32)*(HH/16)*256 * 4);
  unsigned int*   W2pkL = (unsigned int*)  take((size_t)(HH/32)*(HH/16)*256 * 4);
  unsigned short* hhi   = (unsigned short*)take((size_t)BN * FF * 2);   // 20.48 MB
  unsigned short* hlo   = (unsigned short*)take((size_t)BN * FF * 2);   // 20.48 MB
  // y1 bf16 splits alias xw/agg (exactly 80000*256*2 = 80000*128*4 bytes, dead by then)
  unsigned short* y1hi = (unsigned short*)xw;
  unsigned short* y1lo = (unsigned short*)agg;

  // 1) zero atomic targets (graph-capturable memsets)
  hipMemsetAsync(deg, 0, (size_t)BN * 4, stream);
  hipMemsetAsync(agg, 0, (size_t)BN * FF * 4, stream);

  // 2) pre-pack weights into WMMA fragment layouts
  k_pack_b_f32<<<((FF/4)*(FF/16)*64 + 255) / 256, 256, 0, stream>>>(Wc, WcPk, FF, FF);
  k_pack_w_bf16<<<((FF/32)*(HH/16)*256 + 255) / 256, 256, 0, stream>>>(W1, W1pkH, W1pkL, FF, HH);
  k_pack_w_bf16<<<((HH/32)*(HH/16)*256 + 255) / 256, 256, 0, stream>>>(W2, W2pkH, W2pkL, HH, HH);

  // 3) degrees -> dinv
  k_degree<<<(BB * EE + 255) / 256, 256, 0, stream>>>(ei, deg);
  k_dinv<<<(BN + 255) / 256, 256, 0, stream>>>(deg, dinv);

  // 4) xw = x @ Wc  (exact fp32 WMMA)
  {
    int blocks = (BN / 128) * (FF / 64);         // 625 * 2 = 1250
    k_gemm_f32wmma<<<blocks, 256, 0, stream>>>(x, WcPk, xw, BN, FF);
  }

  // 5) edge scatter-add (one wave per edge)
  k_aggregate<<<(BB * EE + 7) / 8, 256, 0, stream>>>(ei, xw, dinv, agg);

  // 6) h = relu(conv) + x0, split to bf16
  {
    size_t n = (size_t)BN * FF;
    k_finalize<<<(unsigned)((n + 255) / 256), 256, 0, stream>>>(agg, xw, dinv, bc,
                                                                x, hhi, hlo);
  }

  // 7) y1 = leaky_relu(h @ W1 + b1), stored as bf16 split
  {
    int blocks = (BN / 128) * (HH / 64);         // 625 * 4 = 2500
    k_gemm_bf16x3<2, FF / 32><<<blocks, 256, 0, stream>>>(hhi, hlo, W1pkH, W1pkL,
                                                          b1, nullptr, y1hi, y1lo,
                                                          BN, HH);
  }

  // 8) out = leaky_relu(y1 @ W2 + b2), fp32
  {
    int blocks = (BN / 128) * (HH / 64);         // 625 * 4 = 2500
    k_gemm_bf16x3<1, HH / 32><<<blocks, 256, 0, stream>>>(y1hi, y1lo, W2pkH, W2pkL,
                                                          b2, out, nullptr, nullptr,
                                                          BN, HH);
  }
}
